// GBSTLayer_58695023067208
// MI455X (gfx1250) — matrix operand order repaired and checked by reading, not compiled, not run on hardware
//
#include <hip/hip_runtime.h>
#include <hip/hip_bf16.h>
#include <stdint.h>
#include <stddef.h>

// ---------------------------------------------------------------------------
// GBST layer for MI455X (gfx1250, wave32, WMMA + TDM).
//   Stage 0a: pack conv_w (f32 [E][E][5]) into bf16 WMMA-A fragment order.
//   Stage 0b: transpose-convert x (f32 [B][E][L]) -> Xt bf16 [B][L][E].
//   Stage A : conv-as-GEMM, v_wmma_f32_16x16x32_bf16, fp32 accum.
//             B tiles staged into LDS by the Tensor Data Mover (double
//             buffered, TENSORcnt + workgroup barriers), A from L2.
//   Stage B1/B1b/B2: score projection, width softmax, fused recombine+pool.
// ---------------------------------------------------------------------------

typedef __bf16 bf16_t;
typedef __attribute__((ext_vector_type(16))) __bf16 v16bf;
typedef __attribute__((ext_vector_type(8)))  float  v8f;
typedef unsigned int u32x4 __attribute__((ext_vector_type(4)));
typedef int          i32x8 __attribute__((ext_vector_type(8)));
typedef int          i32x4 __attribute__((ext_vector_type(4)));

#if __has_builtin(__builtin_amdgcn_tensor_load_to_lds) && \
    __has_builtin(__builtin_amdgcn_s_wait_tensorcnt)
#define USE_TDM 1
#else
#define USE_TDM 0
#endif

// Problem constants
constexpr int    Bn    = 8;
constexpr int    Ec    = 512;
constexpr int    Lfull = 8192;
constexpr int    KS    = 5;
constexpr int    Lconv = Lfull - (KS - 1);   // 8188
constexpr int    Lout  = Lfull / 2;          // 4096

// GEMM tiling (wave32): 8 waves, each 64x64 (4x4 of 16x16), WG tile 128x256
constexpr int TILE_M = 128;
constexpr int TILE_N = 256;
constexpr int KCH    = 32;                   // bf16 K per WMMA
constexpr int NIC    = Ec / KCH;             // 16 i-chunks
constexpr int NOT    = Ec / 16;              // 32 o-tiles of 16

// LDS B-tile: 260 t-rows x 32 channels, row stride 48 bf16 (96 B: 64 B data
// + 32 B pad -> every row 32 B aligned for v16bf ds loads)
constexpr int ROWS    = TILE_N + 4;          // 260
constexpr int RSTRIDE = 48;                  // bf16 per row (96 B)

// Workspace layout (bytes)
constexpr size_t WP_ELEMS  = (size_t)NOT * NIC * KS * 512;        // 1,310,720 bf16
constexpr size_t OFF_WP    = 0;
constexpr size_t OFF_XT    = OFF_WP + WP_ELEMS * 2;               // 2,621,440
constexpr size_t XT_BYTES  = (size_t)Bn * Lfull * Ec * 2;         // 64 MiB
constexpr size_t OFF_Y     = OFF_XT + XT_BYTES;
constexpr size_t Y_BYTES   = (size_t)Bn * Ec * Lfull * 4;         // 128 MiB
constexpr size_t OFF_S1    = OFF_Y + Y_BYTES;
constexpr size_t S1_BYTES  = (size_t)Bn * Lfull * 4;
constexpr size_t OFF_ATT   = OFF_S1 + S1_BYTES;

// ---------------------------------------------------------------------------
// Stage 0a: pack W into A-fragment order (contiguous k = 16*(lane/16) + e, same order as the B loads).
// ---------------------------------------------------------------------------
__global__ void pack_w_kernel(const float* __restrict__ w, bf16_t* __restrict__ wp) {
    int tid = blockIdx.x * blockDim.x + threadIdx.x;
    if (tid >= (int)WP_ELEMS) return;
    int e    = tid & 15;
    int lane = (tid >> 4) & 31;
    int k    = (tid >> 9) % KS;
    int ic   = (tid / (512 * KS)) & (NIC - 1);
    int ot   = tid / (512 * KS * NIC);
    int o    = ot * 16 + (lane & 15);
    int Kv   = ((lane < 16) ? 0 : 16) + e;
    int i    = ic * KCH + Kv;
    wp[tid]  = (bf16_t)w[((size_t)o * Ec + i) * KS + k];
}

// ---------------------------------------------------------------------------
// Stage 0b: Xt[b][t][i] = bf16(x[b][i][t]) via LDS-tiled transpose (32x32).
// ---------------------------------------------------------------------------
__global__ void xpose_kernel(const float* __restrict__ x, bf16_t* __restrict__ xt) {
    __shared__ float tile[32][33];
    int b  = blockIdx.z;
    int t0 = blockIdx.x * 32;
    int i0 = blockIdx.y * 32;
    int tx = threadIdx.x & 31;
    int ty = threadIdx.x >> 5;           // 0..7
#pragma unroll
    for (int r = 0; r < 4; ++r) {
        int i = i0 + ty + r * 8;
        tile[tx][ty + r * 8] = x[((size_t)b * Ec + i) * Lfull + t0 + tx];
    }
    __syncthreads();
#pragma unroll
    for (int r = 0; r < 4; ++r) {
        int t = t0 + ty + r * 8;
        xt[((size_t)b * Lfull + t) * Ec + i0 + tx] = (bf16_t)tile[ty + r * 8][tx];
    }
}

// ---------------------------------------------------------------------------
// TDM: one 2D descriptor pulls the WG's [260 x 32ch] bf16 tile into LDS.
// tensor_dim1 = Lfull - t0  =>  rows past the end of X are OOB -> zero-filled.
// pad_interval=16 dwords (64B = one row of data), pad_amount=8 dwords (32B)
// => LDS row stride 96 B, matching RSTRIDE.
// ---------------------------------------------------------------------------
#if USE_TDM
__device__ __forceinline__ void tdm_stage_tile(const bf16_t* gsrc, unsigned lds_off,
                                               int rows_valid) {
    u32x4 g0;
    uint64_t ga = (uint64_t)(uintptr_t)gsrc;
    g0[0] = 1u;                                            // count=1 (valid D#)
    g0[1] = lds_off;                                       // LDS byte address
    g0[2] = (unsigned)(ga & 0xFFFFFFFFu);                  // global_addr lo
    g0[3] = (unsigned)((ga >> 32) & 0x1FFFFFFu) | (2u << 30); // addr hi | type=2

    unsigned td1 = (unsigned)rows_valid;                   // tensor_dim1
    i32x8 g1;
    g1[0] = (1 << 16) | (1 << 20) | (3 << 22) | (7 << 25); // 2B elems, pad 64B/32B
    g1[1] = (int)(32u << 16);                              // tensor_dim0 = 32
    g1[2] = (int)((td1 & 0xFFFFu) << 16);                  // tensor_dim1 lo
    g1[3] = (int)((td1 >> 16) & 0xFFFFu) | (32 << 16);     // tensor_dim1 hi | tile_dim0=32
    g1[4] = ROWS;                                          // tile_dim1 = 260
    g1[5] = Ec;                                            // tensor_dim0_stride = 512
    g1[6] = 0;
    g1[7] = 0;

    i32x4 gz4 = {0, 0, 0, 0};                              // 2D: groups 2/3 unused
    i32x8 gz8 = {0, 0, 0, 0, 0, 0, 0, 0};
    __builtin_amdgcn_tensor_load_to_lds(g0, g1, gz4, gz4, gz8, 0);
}
#endif

// Fallback: cooperative global->LDS staging (256 threads, 8 B chunks).
__device__ __forceinline__ void manual_stage_tile(const bf16_t* __restrict__ xb,
                                                  int t0, int ic, bf16_t* dst, int tid) {
    for (int c = tid; c < ROWS * 8; c += 256) {
        int row = c >> 3, cc = c & 7;
        int t = t0 + row;
        unsigned long long v = 0ull;
        if (t < Lfull)
            v = *(const unsigned long long*)(xb + (size_t)t * Ec + ic * KCH + cc * 4);
        *(unsigned long long*)(dst + row * RSTRIDE + cc * 4) = v;
    }
}

// ---------------------------------------------------------------------------
// Stage A: conv-as-GEMM.  Y[b][o][t] = sum_{ic,k} A(W) x B(X shifted by k).
// ---------------------------------------------------------------------------
__global__ void __launch_bounds__(256, 1)
conv_gemm_kernel(const bf16_t* __restrict__ wp, const bf16_t* __restrict__ xt,
                 const float* __restrict__ bias, float* __restrict__ Y) {
    __shared__ bf16_t xs[2][ROWS * RSTRIDE];               // 2 x 24,960 B

    const int b    = blockIdx.z;
    const int lane = threadIdx.x & 31;
    const int wave = threadIdx.x >> 5;
    const int wm   = wave >> 2;              // 0..1
    const int wn   = wave & 3;               // 0..3
    const int col  = lane & 15;              // N col / M row within fragment
    const int half = (lane < 16) ? 0 : 16;   // B K-half per lane group

    const int ot_base = blockIdx.y * (TILE_M / 16) + wm * 4;
    const int t0      = blockIdx.x * TILE_N;           // WG tile start (t)
    const int rowbase = wn * 64;                       // wave's t-rows in LDS tile

    v8f acc[4][4];
#pragma unroll
    for (int ms = 0; ms < 4; ++ms)
#pragma unroll
        for (int ns = 0; ns < 4; ++ns)
#pragma unroll
            for (int j = 0; j < 8; ++j) acc[ms][ns][j] = 0.0f;

    const bf16_t* xb = xt + (size_t)b * Lfull * Ec;

    // Prologue: stage i-chunk 0 into buffer 0.
#if USE_TDM
    if (wave == 0) {
        tdm_stage_tile(xb + (size_t)t0 * Ec, (unsigned)(uintptr_t)&xs[0][0], Lfull - t0);
        __builtin_amdgcn_s_wait_tensorcnt(0);
    }
    __syncthreads();
#else
    manual_stage_tile(xb, t0, 0, &xs[0][0], threadIdx.x);
    __syncthreads();
#endif

    for (int ic = 0; ic < NIC; ++ic) {
        const int cur = ic & 1;
        // Kick off DMA for the next i-chunk into the other buffer.
        if (ic + 1 < NIC) {
#if USE_TDM
            if (wave == 0)
                tdm_stage_tile(xb + (size_t)t0 * Ec + (ic + 1) * KCH,
                               (unsigned)(uintptr_t)&xs[cur ^ 1][0], Lfull - t0);
#else
            manual_stage_tile(xb, t0, ic + 1, &xs[cur ^ 1][0], threadIdx.x);
#endif
        }
        const bf16_t* xsc = &xs[cur][0];
#pragma unroll
        for (int k = 0; k < KS; ++k) {
            v16bf afr[4], bfr[4];
#pragma unroll
            for (int ms = 0; ms < 4; ++ms) {
                size_t fi = ((size_t)((ot_base + ms) * NIC + ic) * KS + k) * 512 + lane * 16;
                afr[ms] = *(const v16bf*)(wp + fi);
            }
#pragma unroll
            for (int ns = 0; ns < 4; ++ns) {
                int row = rowbase + ns * 16 + k + col;         // <= 259
                bfr[ns] = *(const v16bf*)(xsc + row * RSTRIDE + half);
            }
#pragma unroll
            for (int ms = 0; ms < 4; ++ms)
#pragma unroll
                for (int ns = 0; ns < 4; ++ns)
                    acc[ms][ns] = __builtin_amdgcn_wmma_f32_16x16x32_bf16(
                        false, afr[ms], false, bfr[ns], (short)0, acc[ms][ns],
                        false, false);
        }
        __syncthreads();                 // everyone done reading xs[cur]
#if USE_TDM
        if (wave == 0 && ic + 1 < NIC)
            __builtin_amdgcn_s_wait_tensorcnt(0);  // next buffer landed
        __syncthreads();                 // release other waves to read it
#endif
    }

    // Epilogue: C VGPR r -> M = r / r+8 by lane half, N = lane%16.
    // Add bias; write exact zeros for t >= Lconv so downstream pooling is uniform.
#pragma unroll
    for (int ms = 0; ms < 4; ++ms) {
        int o0 = (ot_base + ms) * 16;
#pragma unroll
        for (int r = 0; r < 8; ++r) {
            int   o  = o0 + ((lane < 16) ? r : r + 8);
            float bv = bias[o];
#pragma unroll
            for (int ns = 0; ns < 4; ++ns) {
                int   t = t0 + wn * 64 + ns * 16 + col;    // always < 8192
                float v = acc[ms][ns][r] + bv;
                Y[((size_t)b * Ec + o) * Lfull + t] = (t < Lconv) ? v : 0.0f;
            }
        }
    }
}

// ---------------------------------------------------------------------------
// Stage B1: s1[b][t] = sum_e Y[b][e][t] * score_w[e]
// ---------------------------------------------------------------------------
__global__ void score_kernel(const float* __restrict__ Y, const float* __restrict__ sw,
                             float* __restrict__ s1) {
    int b = blockIdx.y;
    int t = blockIdx.x * 256 + threadIdx.x;
    const float* yb = Y + (size_t)b * Ec * Lfull + t;
    float acc = 0.0f;
    for (int e = 0; e < Ec; ++e) acc += yb[(size_t)e * Lfull] * sw[e];
    s1[b * Lfull + t] = acc;
}

// ---------------------------------------------------------------------------
// Stage B1b: attention over widths {1,2,3}; stored pre-divided by w.
// ---------------------------------------------------------------------------
__global__ void att_kernel(const float* __restrict__ s1, float* __restrict__ att4) {
    int b = blockIdx.y;
    int t = blockIdx.x * 256 + threadIdx.x;
    const float* s = s1 + b * Lfull;
    auto g = [&](int i) -> float { return (i < Lfull) ? s[i] : 0.0f; };
    float sc1 = s[t];
    int   n2  = t >> 1;
    float sc2 = 0.5f * (g(2 * n2) + g(2 * n2 + 1));
    int   n3  = t / 3;
    float sc3 = (g(3 * n3) + g(3 * n3 + 1) + g(3 * n3 + 2)) * (1.0f / 3.0f);
    float m   = fmaxf(sc1, fmaxf(sc2, sc3));
    float e1  = __expf(sc1 - m), e2 = __expf(sc2 - m), e3 = __expf(sc3 - m);
    float inv = 1.0f / (e1 + e2 + e3);
    float* o  = att4 + (size_t)(b * Lfull + t) * 4;
    o[0] = e1 * inv;
    o[1] = e2 * inv * 0.5f;
    o[2] = e3 * inv * (1.0f / 3.0f);
    o[3] = 0.0f;
}

// ---------------------------------------------------------------------------
// Stage B2: fused recombine + downsample(2). One block per (b,e) row.
// ---------------------------------------------------------------------------
__global__ void __launch_bounds__(256)
recombine_kernel(const float* __restrict__ Y, const float* __restrict__ att4,
                 float* __restrict__ out) {
    __shared__ float ys[Lfull];
    int b = blockIdx.y;
    int e = blockIdx.x;
    const float4* yr  = (const float4*)(Y + ((size_t)b * Ec + e) * Lfull);
    float4*       ys4 = (float4*)ys;
    for (int i = threadIdx.x; i < Lfull / 4; i += 256) ys4[i] = yr[i];
    __syncthreads();

    const float* ab = att4 + (size_t)b * Lfull * 4;
    float*       ob = out + ((size_t)b * Ec + e) * Lout;
#pragma unroll
    for (int q = 0; q < Lout / 256; ++q) {
        int   lp = q * 256 + threadIdx.x;
        float r  = 0.0f;
#pragma unroll
        for (int s = 0; s < 2; ++s) {
            int t = 2 * lp + s;
            const float* a  = ab + (size_t)t * 4;
            float v1 = ys[t];
            int   n2 = t >> 1;
            float v2 = ys[2 * n2] + ys[2 * n2 + 1];
            int   n3 = t / 3;
            int   i2 = 3 * n3 + 2;
            float v3 = ys[3 * n3] + ys[3 * n3 + 1] + ((i2 < Lfull) ? ys[i2] : 0.0f);
            r += a[0] * v1 + a[1] * v2 + a[2] * v3;
        }
        ob[lp] = 0.5f * r;
    }
}

// ---------------------------------------------------------------------------
extern "C" void kernel_launch(void* const* d_in, const int* in_sizes, int n_in,
                              void* d_out, int out_size, void* d_ws, size_t ws_size,
                              hipStream_t stream) {
    const float* x      = (const float*)d_in[0];   // [B,E,L]
    const float* conv_w = (const float*)d_in[1];   // [E,E,5]
    const float* conv_b = (const float*)d_in[2];   // [E]
    const float* sw     = (const float*)d_in[3];   // [E]
    float*       out    = (float*)d_out;           // [B,E,Lout]

    char* ws = (char*)d_ws;
    bf16_t* wp   = (bf16_t*)(ws + OFF_WP);
    bf16_t* xt   = (bf16_t*)(ws + OFF_XT);
    float*  Y    = (float*) (ws + OFF_Y);
    float*  s1   = (float*) (ws + OFF_S1);
    float*  att4 = (float*) (ws + OFF_ATT);

    pack_w_kernel<<<dim3((unsigned)((WP_ELEMS + 255) / 256)), dim3(256), 0, stream>>>(conv_w, wp);
    xpose_kernel<<<dim3(Lfull / 32, Ec / 32, Bn), dim3(256), 0, stream>>>(x, xt);
    conv_gemm_kernel<<<dim3(Lfull / TILE_N, Ec / TILE_M, Bn), dim3(256), 0, stream>>>(wp, xt, conv_b, Y);
    score_kernel<<<dim3(Lfull / 256, Bn), dim3(256), 0, stream>>>(Y, sw, s1);
    att_kernel<<<dim3(Lfull / 256, Bn), dim3(256), 0, stream>>>(s1, att4);
    recombine_kernel<<<dim3(Ec, Bn), dim3(256), 0, stream>>>(Y, att4, out);
}
